// SNNLoss_21457656611487
// MI455X (gfx1250) — matrix operand order repaired and checked by reading, not compiled, not run on hardware
//
#include <hip/hip_runtime.h>
#include <hip/hip_bf16.h>

typedef __attribute__((ext_vector_type(16))) _Float16 v16h;
typedef __attribute__((ext_vector_type(8)))  float    v8f;

#define BATCH  2048
#define DD     32
#define JSPLIT 8
#define JSLICE (BATCH / JSPLIT)   // 256 j's per slice
#define NCHUNK (JSLICE / 32)      // 8 chunks of 32 j's

// One wave (32 lanes) per block. Each block owns a 16-row i-tile and one
// j-slice of 256 columns. For each 32-j chunk it computes the 16x32 tile of
// ed0 and edsum, then accumulates class-binned row sums with
// v_wmma_f32_16x16x32_f16:  C[i, c] += sum_j ed[i,j] * onehot(y_j)[c].
__global__ __launch_bounds__(32)
void snn_tile_kernel(const float* __restrict__ x, const int* __restrict__ y,
                     float* __restrict__ ws_num, float* __restrict__ ws_den,
                     float* __restrict__ ws_den1)
{
    __shared__ float sxj[32 * DD];   // 32 staged x_j rows (4 KB)
    __shared__ int   syj[32];        // staged y_j

    const int lane  = threadIdx.x;   // 0..31
    const int il    = lane & 15;     // A-row (i) for this lane == C column (class)
    const int h     = lane >> 4;     // half-wave select (WMMA K split)
    const int itile = blockIdx.x;
    const int jsl   = blockIdx.y;
    const int gi    = itile * 16 + il;

    // x_i row pinned in VGPRs (rows duplicated across the two half-waves; cached)
    float xi[DD];
    {
        const float4* xr = (const float4*)(x + gi * DD);
        #pragma unroll
        for (int q = 0; q < DD / 4; ++q) {
            float4 v = xr[q];
            xi[4*q+0] = v.x; xi[4*q+1] = v.y; xi[4*q+2] = v.z; xi[4*q+3] = v.w;
        }
    }

    v8f cnum = {};   // C[i, c] accum of ed0
    v8f csum = {};   // C[i, c] accum of edsum (k = 1..31)

    for (int c = 0; c < NCHUNK; ++c) {
        const int j0 = jsl * JSLICE + c * 32;

        // cooperative stage: one x_j row + one y_j per lane into LDS
        {
            const float4* src = (const float4*)(x + (j0 + lane) * DD);
            float4*       dst = (float4*)(sxj + lane * DD);
            #pragma unroll
            for (int q = 0; q < DD / 4; ++q) dst[q] = src[q];
            syj[lane] = y[j0 + lane];
        }
        __syncthreads();

        // Build A (ed0), A' (edsum), B (one-hot) in the wave32 WMMA layouts.
        v16h a0, a1, bmat;
        #pragma unroll
        for (int e = 0; e < 16; ++e) {
            // A element e of this lane covers contraction index K:
            //   lanes 0-15: K = 0..7, 16..23 ; lanes 16-31: K = 8..15, 24..31
            const int K  = (e < 8) ? (8 * h + e) : (8 * h + e + 8);
            const int gj = j0 + K;
            const float* xj = sxj + K * DD;

            float d0 = xi[0] - xj[0];
            float e0 = __expf(-d0 * d0);          // T = 1.0
            float s  = 0.f;
            #pragma unroll
            for (int k = 1; k < DD; ++k) {
                float dk = xi[k] - xj[k];
                s += __expf(-dk * dk);
            }
            const bool diag = (gj == gi);         // off_diag mask, branchless
            a0[e] = (_Float16)(diag ? 0.f : e0);
            a1[e] = (_Float16)(diag ? 0.f : s);

            // B element e covers row K (same as A), column (class) = il
            bmat[e] = (_Float16)((syj[K] == il) ? 1.f : 0.f);
        }

        cnum = __builtin_amdgcn_wmma_f32_16x16x32_f16(false, a0, false, bmat,
                                                      (short)0, cnum, false, false);
        csum = __builtin_amdgcn_wmma_f32_16x16x32_f16(false, a1, false, bmat,
                                                      (short)0, csum, false, false);
        __syncthreads();   // protect LDS re-stage
    }

    // Spill the two 16x16 f32 C tiles to LDS and extract per-i partials.
    float* snum = sxj;         // 256 floats
    float* ssum = sxj + 256;   // 256 floats
    #pragma unroll
    for (int r = 0; r < 8; ++r) {
        const int M = (h == 0) ? r : (r + 8);    // C layout: M = r + 8*h
        snum[M * 16 + il] = cnum[r];
        ssum[M * 16 + il] = csum[r];
    }
    __syncthreads();

    if (lane < 16) {
        const int i   = lane;
        const int g   = itile * 16 + i;
        const int cls = y[g];
        float num = snum[i * 16 + cls];           // sum_j ed0 * same
        float den = 0.f;                          // sum_j ed0 (all classes)
        #pragma unroll
        for (int cc = 0; cc < 16; ++cc) den += snum[i * 16 + cc];
        float den1 = ssum[i * 16 + cls] * (1.0f / (float)(DD - 1));

        const int idx = jsl * BATCH + g;          // unique per (slice, i): deterministic
        ws_num [idx] = num;
        ws_den [idx] = den;
        ws_den1[idx] = den1;
    }
}

__global__ __launch_bounds__(256)
void snn_reduce_kernel(const float* __restrict__ ws_num,
                       const float* __restrict__ ws_den,
                       const float* __restrict__ ws_den1,
                       float* __restrict__ out)
{
    __shared__ float red[256];
    const int t = threadIdx.x;
    float acc = 0.f;
    for (int i = t; i < BATCH; i += 256) {
        float num = 0.f, den = 0.f, den1 = 0.f;
        #pragma unroll
        for (int s = 0; s < JSPLIT; ++s) {
            num  += ws_num [s * BATCH + i];
            den  += ws_den [s * BATCH + i];
            den1 += ws_den1[s * BATCH + i];
        }
        acc += -logf(1e-5f + num / (1e-5f + 0.5f * den + 0.5f * den1));
    }
    red[t] = acc;
    __syncthreads();
    for (int s = 128; s > 0; s >>= 1) {
        if (t < s) red[t] += red[t + s];
        __syncthreads();
    }
    if (t == 0) out[0] = red[0] / (float)BATCH;
}

extern "C" void kernel_launch(void* const* d_in, const int* in_sizes, int n_in,
                              void* d_out, int out_size, void* d_ws, size_t ws_size,
                              hipStream_t stream) {
    (void)in_sizes; (void)n_in; (void)out_size; (void)ws_size;
    const float* x = (const float*)d_in[0];
    const int*   y = (const int*)d_in[1];

    float* ws_num  = (float*)d_ws;                       // JSPLIT * BATCH floats
    float* ws_den  = ws_num + (size_t)JSPLIT * BATCH;
    float* ws_den1 = ws_den + (size_t)JSPLIT * BATCH;

    dim3 grid(BATCH / 16, JSPLIT);   // 128 i-tiles x 8 j-slices, 1 wave each
    snn_tile_kernel<<<grid, 32, 0, stream>>>(x, y, ws_num, ws_den, ws_den1);
    snn_reduce_kernel<<<1, 256, 0, stream>>>(ws_num, ws_den, ws_den1, (float*)d_out);
}
